// PointNetFeaturePropagation_26620207301336
// MI455X (gfx1250) — compile-verified, matrix operand order checked by reading, not run on hardware
//
#include <hip/hip_runtime.h>
#include <hip/hip_bf16.h>
#include <stdint.h>

// ---------------------------------------------------------------------------
// Problem constants (match the reference)
// ---------------------------------------------------------------------------
#define B_    4
#define N_    8192
#define S_    2048
#define C1_   128
#define C2_   256
#define CIN_  384
#define H_    256
#define JTOT  (B_ * N_)        // 32768 columns (b-major, then n)
#define KT1   (CIN_ / 32)      // 12 K-tiles for GEMM1
#define KT2   (H_   / 32)      // 8  K-tiles for GEMM2
#define JT_   (JTOT / 16)      // 2048 column tiles
#define MT_   (H_   / 16)      // 16 row tiles
#define FRAG_BYTES 1024        // one 16x32 (or 32x16) bf16 WMMA fragment

typedef __attribute__((ext_vector_type(16))) __bf16 v16bf;
typedef __attribute__((ext_vector_type(8)))  float  v8f;

// round-to-nearest-even fp32 -> bf16
__device__ __forceinline__ unsigned short f2bf(float f) {
    union { float f; unsigned u; } x; x.f = f;
    unsigned r = x.u + 0x7FFFu + ((x.u >> 16) & 1u);
    return (unsigned short)(r >> 16);
}

// K-offset within a 32-wide K tile for element e (0..15) of lane `lane`,
// per CDNA5 ISA 7.12.2 16-bit A/B fragment layout.
__device__ __forceinline__ int koff(int lane, int e) {
    return ((lane >> 4) << 3) + ((e >> 3) << 4) + (e & 7);
}

__device__ __forceinline__ v16bf load_frag(const unsigned char* __restrict__ base, int lane) {
    const uint4* p = (const uint4*)(base + lane * 32);
    union { uint4 u[2]; v16bf v; } x;
    x.u[0] = p[0];
    x.u[1] = p[1];
    return x.v;
}

// ---------------------------------------------------------------------------
// 1) Brute-force 3-NN + inverse-distance weights.  One thread per query
//    point; xyz2 staged through LDS in 256-point tiles.
// ---------------------------------------------------------------------------
__global__ __launch_bounds__(256) void knn3_kernel(
    const float* __restrict__ xyz1, const float* __restrict__ xyz2,
    int* __restrict__ idx, float* __restrict__ wgt)
{
    __shared__ float sx[256], sy[256], sz[256];
    int gid = blockIdx.x * 256 + threadIdx.x;   // b*N + n  (block stays in one b)
    int b = gid >> 13;
    float qx = xyz1[gid * 3 + 0];
    float qy = xyz1[gid * 3 + 1];
    float qz = xyz1[gid * 3 + 2];

    float d0 = 3.0e38f, d1 = 3.0e38f, d2 = 3.0e38f;
    int   i0 = 0, i1 = 0, i2 = 0;

    for (int s0 = 0; s0 < S_; s0 += 256) {
        int sl = s0 + threadIdx.x;
        const float* p = xyz2 + ((size_t)b * S_ + sl) * 3;
        sx[threadIdx.x] = p[0];
        sy[threadIdx.x] = p[1];
        sz[threadIdx.x] = p[2];
        __syncthreads();
        #pragma unroll 4
        for (int t = 0; t < 256; ++t) {
            float dx = qx - sx[t], dy = qy - sy[t], dz = qz - sz[t];
            float d = dx * dx + dy * dy + dz * dz;
            int s = s0 + t;
            if (d < d0)      { d2 = d1; i2 = i1; d1 = d0; i1 = i0; d0 = d; i0 = s; }
            else if (d < d1) { d2 = d1; i2 = i1; d1 = d;  i1 = s; }
            else if (d < d2) { d2 = d;  i2 = s; }
        }
        __syncthreads();
    }

    float r0 = 1.0f / (d0 + 1e-8f);
    float r1 = 1.0f / (d1 + 1e-8f);
    float r2 = 1.0f / (d2 + 1e-8f);
    float inv = 1.0f / (r0 + r1 + r2);
    idx[gid * 3 + 0] = i0;  wgt[gid * 3 + 0] = r0 * inv;
    idx[gid * 3 + 1] = i1;  wgt[gid * 3 + 1] = r1 * inv;
    idx[gid * 3 + 2] = i2;  wgt[gid * 3 + 2] = r2 * inv;
}

// ---------------------------------------------------------------------------
// 2) Pack a [256 x K] fp32 weight matrix into bf16 A-fragments.
//    Fragment fa = mt*KT + kt; lane holds row m = mt*16 + (lane&15).
// ---------------------------------------------------------------------------
__global__ __launch_bounds__(256) void packW_kernel(
    const float* __restrict__ W, unsigned char* __restrict__ dst, int K, int KTloc)
{
    int tid  = blockIdx.x * 256 + threadIdx.x;
    int lane = tid & 31;
    int f    = tid >> 5;
    if (f >= MT_ * KTloc) return;
    int kt = f % KTloc, mt = f / KTloc;
    int m  = mt * 16 + (lane & 15);

    unsigned short v[16];
    #pragma unroll
    for (int e = 0; e < 16; ++e) {
        int k = kt * 32 + koff(lane, e);
        v[e] = f2bf(W[(size_t)m * K + k]);
    }
    unsigned u[8];
    #pragma unroll
    for (int i = 0; i < 8; ++i) u[i] = (unsigned)v[2 * i] | ((unsigned)v[2 * i + 1] << 16);
    uint4* out = (uint4*)(dst + (size_t)f * FRAG_BYTES + lane * 32);
    out[0] = make_uint4(u[0], u[1], u[2], u[3]);
    out[1] = make_uint4(u[4], u[5], u[6], u[7]);
}

// ---------------------------------------------------------------------------
// 3) Build X0 = concat(points1, interp(points2)) as bf16 B-fragments.
//    Fragment fb = jt*KT1 + kt. kt<4 -> points1 rows; kt>=4 -> interpolation.
// ---------------------------------------------------------------------------
__global__ __launch_bounds__(256) void buildX0_kernel(
    const float* __restrict__ points1, const float* __restrict__ points2,
    const int* __restrict__ idx, const float* __restrict__ wgt,
    unsigned char* __restrict__ X0)
{
    int tid  = blockIdx.x * 256 + threadIdx.x;
    int lane = tid & 31;
    int f    = tid >> 5;
    int kt = f % KT1, jt = f / KT1;
    int j  = jt * 16 + (lane & 15);
    int b  = j >> 13, n = j & (N_ - 1);

    unsigned short v[16];
    if (kt < 4) {                     // points1 channels 0..127
        const float* p1 = points1 + ((size_t)b * C1_) * N_ + n;
        #pragma unroll
        for (int e = 0; e < 16; ++e) {
            int c = kt * 32 + koff(lane, e);
            v[e] = f2bf(p1[(size_t)c * N_]);
        }
    } else {                          // interpolated points2 channels
        int   q  = (b * N_ + n) * 3;
        int   a0 = idx[q + 0], a1 = idx[q + 1], a2 = idx[q + 2];
        float w0 = wgt[q + 0], w1 = wgt[q + 1], w2 = wgt[q + 2];
        const float* p2 = points2 + (size_t)b * C2_ * S_;
        #pragma unroll
        for (int e = 0; e < 16; ++e) {
            int c = (kt - 4) * 32 + koff(lane, e);
            const float* row = p2 + (size_t)c * S_;
            v[e] = f2bf(w0 * row[a0] + w1 * row[a1] + w2 * row[a2]);
        }
    }
    unsigned u[8];
    #pragma unroll
    for (int i = 0; i < 8; ++i) u[i] = (unsigned)v[2 * i] | ((unsigned)v[2 * i + 1] << 16);
    uint4* out = (uint4*)(X0 + (size_t)f * FRAG_BYTES + lane * 32);
    out[0] = make_uint4(u[0], u[1], u[2], u[3]);
    out[1] = make_uint4(u[4], u[5], u[6], u[7]);
}

// ---------------------------------------------------------------------------
// 4) WMMA GEMM: Y[256 x 32768] = A[256 x K] * X[K x 32768] (bf16 -> f32).
//    Block = 8 waves; wave tile = 64(M) x 32(N); block tile = 256 x 64.
// ---------------------------------------------------------------------------
__global__ __launch_bounds__(256) void gemm_wmma_kernel(
    const unsigned char* __restrict__ Apack, const unsigned char* __restrict__ Bpack,
    float* __restrict__ Y, int KTloc)
{
    int lane = threadIdx.x & 31;
    int wave = threadIdx.x >> 5;
    int wm   = wave & 3;             // 4 M-groups of 64 rows
    int wn   = wave >> 2;            // 2 N-groups of 32 cols
    int jt0  = blockIdx.x * 4 + wn * 2;

    v8f acc[4][2];
    #pragma unroll
    for (int m = 0; m < 4; ++m) { acc[m][0] = (v8f)(0.0f); acc[m][1] = (v8f)(0.0f); }

    const unsigned char* bcol0 = Bpack + (size_t)(jt0 + 0) * KTloc * FRAG_BYTES;
    const unsigned char* bcol1 = Bpack + (size_t)(jt0 + 1) * KTloc * FRAG_BYTES;
    const unsigned char* arow  = Apack + (size_t)(wm * 4) * KTloc * FRAG_BYTES;

    for (int kt = 0; kt < KTloc; ++kt) {
        __builtin_prefetch(bcol0 + (size_t)(kt + 1) * FRAG_BYTES + lane * 32, 0, 1);
        __builtin_prefetch(bcol1 + (size_t)(kt + 1) * FRAG_BYTES + lane * 32, 0, 1);
        v16bf bf0 = load_frag(bcol0 + (size_t)kt * FRAG_BYTES, lane);
        v16bf bf1 = load_frag(bcol1 + (size_t)kt * FRAG_BYTES, lane);
        #pragma unroll
        for (int m = 0; m < 4; ++m) {
            v16bf af = load_frag(arow + (size_t)(m * KTloc + kt) * FRAG_BYTES, lane);
            acc[m][0] = __builtin_amdgcn_wmma_f32_16x16x32_bf16(
                false, af, false, bf0, (short)0, acc[m][0], false, false);
            acc[m][1] = __builtin_amdgcn_wmma_f32_16x16x32_bf16(
                false, af, false, bf1, (short)0, acc[m][1], false, false);
        }
    }

    // C/D layout: VGPR r -> row r + 8*(lane>=16); col = lane&15
    int rbase = 8 * (lane >> 4);
    int col   = lane & 15;
    #pragma unroll
    for (int m = 0; m < 4; ++m) {
        int otile = (wm * 4 + m) * 16;
        #pragma unroll
        for (int t = 0; t < 2; ++t) {
            int j = (jt0 + t) * 16 + col;
            #pragma unroll
            for (int r = 0; r < 8; ++r) {
                Y[(size_t)(otile + rbase + r) * JTOT + j] = acc[m][t][r];
            }
        }
    }
}

// ---------------------------------------------------------------------------
// 5) BatchNorm statistics: one block per channel; produce scale/shift.
// ---------------------------------------------------------------------------
__global__ __launch_bounds__(256) void bn_stats_kernel(
    const float* __restrict__ Y, const float* __restrict__ gamma,
    const float* __restrict__ beta, float* __restrict__ scale,
    float* __restrict__ shift)
{
    __shared__ float sh[256], sh2[256];
    int o = blockIdx.x, t = threadIdx.x;
    const float* row = Y + (size_t)o * JTOT;
    float s = 0.0f, s2 = 0.0f;
    for (int j = t; j < JTOT; j += 256) {
        float v = row[j];
        s += v; s2 += v * v;
    }
    sh[t] = s; sh2[t] = s2;
    __syncthreads();
    for (int off = 128; off > 0; off >>= 1) {
        if (t < off) { sh[t] += sh[t + off]; sh2[t] += sh2[t + off]; }
        __syncthreads();
    }
    if (t == 0) {
        float mean = sh[0] * (1.0f / JTOT);
        float var  = sh2[0] * (1.0f / JTOT) - mean * mean;
        float sc   = gamma[o] * rsqrtf(var + 1e-5f);
        scale[o] = sc;
        shift[o] = beta[o] - mean * sc;
    }
}

// ---------------------------------------------------------------------------
// 6) BN + ReLU + bf16 repack of Y into B-fragments for GEMM2.
// ---------------------------------------------------------------------------
__global__ __launch_bounds__(256) void bnrelu_pack_kernel(
    const float* __restrict__ Y, const float* __restrict__ scale,
    const float* __restrict__ shift, unsigned char* __restrict__ X1)
{
    int tid  = blockIdx.x * 256 + threadIdx.x;
    int lane = tid & 31;
    int f    = tid >> 5;
    int kt = f % KT2, jt = f / KT2;
    int j  = jt * 16 + (lane & 15);

    unsigned short v[16];
    #pragma unroll
    for (int e = 0; e < 16; ++e) {
        int k = kt * 32 + koff(lane, e);
        float y = Y[(size_t)k * JTOT + j];
        float r = fmaxf(0.0f, scale[k] * y + shift[k]);
        v[e] = f2bf(r);
    }
    unsigned u[8];
    #pragma unroll
    for (int i = 0; i < 8; ++i) u[i] = (unsigned)v[2 * i] | ((unsigned)v[2 * i + 1] << 16);
    uint4* out = (uint4*)(X1 + (size_t)f * FRAG_BYTES + lane * 32);
    out[0] = make_uint4(u[0], u[1], u[2], u[3]);
    out[1] = make_uint4(u[4], u[5], u[6], u[7]);
}

// ---------------------------------------------------------------------------
// 7) Final BN + ReLU -> d_out [B, 256, N] fp32.
// ---------------------------------------------------------------------------
__global__ __launch_bounds__(256) void final_bnrelu_kernel(
    const float* __restrict__ Y, const float* __restrict__ scale,
    const float* __restrict__ shift, float* __restrict__ out)
{
    int tid = blockIdx.x * 256 + threadIdx.x;   // == (b*256 + o)*N + n
    int n = tid & (N_ - 1);
    int o = (tid >> 13) & (H_ - 1);
    int b = tid >> 21;
    int j = b * N_ + n;
    float y = Y[(size_t)o * JTOT + j];
    out[tid] = fmaxf(0.0f, scale[o] * y + shift[o]);
}

// ---------------------------------------------------------------------------
// Launcher
// ---------------------------------------------------------------------------
extern "C" void kernel_launch(void* const* d_in, const int* in_sizes, int n_in,
                              void* d_out, int out_size, void* d_ws, size_t ws_size,
                              hipStream_t stream)
{
    (void)in_sizes; (void)n_in; (void)out_size; (void)ws_size;
    const float* xyz1    = (const float*)d_in[0];
    const float* xyz2    = (const float*)d_in[1];
    const float* points1 = (const float*)d_in[2];
    const float* points2 = (const float*)d_in[3];
    const float* W1      = (const float*)d_in[4];
    const float* gamma1  = (const float*)d_in[6];
    const float* beta1   = (const float*)d_in[7];
    const float* W2      = (const float*)d_in[8];
    const float* gamma2  = (const float*)d_in[10];
    const float* beta2   = (const float*)d_in[11];
    float* out = (float*)d_out;

    // workspace layout (all 1 KB aligned)
    unsigned char* ws = (unsigned char*)d_ws;
    size_t off = 0;
    auto take = [&](size_t bytes) {
        unsigned char* p = ws + off;
        off += (bytes + 1023) & ~(size_t)1023;
        return p;
    };
    int*           idx   = (int*)  take((size_t)JTOT * 3 * sizeof(int));
    float*         wgt   = (float*)take((size_t)JTOT * 3 * sizeof(float));
    unsigned char* w1p   = take((size_t)MT_ * KT1 * FRAG_BYTES);       // 192 KB
    unsigned char* w2p   = take((size_t)MT_ * KT2 * FRAG_BYTES);       // 128 KB
    float*         sc1   = (float*)take(H_ * sizeof(float));
    float*         sh1   = (float*)take(H_ * sizeof(float));
    float*         sc2   = (float*)take(H_ * sizeof(float));
    float*         sh2   = (float*)take(H_ * sizeof(float));
    unsigned char* X0    = take((size_t)JT_ * KT1 * FRAG_BYTES);       // 24 MB
    unsigned char* X1    = take((size_t)JT_ * KT2 * FRAG_BYTES);       // 16 MB
    float*         Ybuf  = (float*)take((size_t)H_ * JTOT * sizeof(float)); // 32 MB (reused)

    // 1) 3-NN + weights
    knn3_kernel<<<JTOT / 256, 256, 0, stream>>>(xyz1, xyz2, idx, wgt);

    // 2) pack weights into WMMA A-fragments
    packW_kernel<<<(MT_ * KT1 * 32 + 255) / 256, 256, 0, stream>>>(W1, w1p, CIN_, KT1);
    packW_kernel<<<(MT_ * KT2 * 32 + 255) / 256, 256, 0, stream>>>(W2, w2p, H_, KT2);

    // 3) build interpolated/concatenated activations as B-fragments
    buildX0_kernel<<<(JT_ * KT1 * 32) / 256, 256, 0, stream>>>(points1, points2, idx, wgt, X0);

    // 4) GEMM1 -> Y
    gemm_wmma_kernel<<<JT_ / 4, 256, 0, stream>>>(w1p, X0, Ybuf, KT1);

    // 5) BN stats layer 1
    bn_stats_kernel<<<H_, 256, 0, stream>>>(Ybuf, gamma1, beta1, sc1, sh1);

    // 6) BN+ReLU+repack -> X1
    bnrelu_pack_kernel<<<(JT_ * KT2 * 32) / 256, 256, 0, stream>>>(Ybuf, sc1, sh1, X1);

    // 7) GEMM2 -> Y (buffer reuse is safe: X1 fully materialized)
    gemm_wmma_kernel<<<JT_ / 4, 256, 0, stream>>>(w2p, X1, Ybuf, KT2);

    // 8) BN stats layer 2
    bn_stats_kernel<<<H_, 256, 0, stream>>>(Ybuf, gamma2, beta2, sc2, sh2);

    // 9) final BN+ReLU -> output [B, 256, N]
    final_bnrelu_kernel<<<(B_ * H_ * N_) / 256, 256, 0, stream>>>(Ybuf, sc2, sh2, out);
}
